// SpikeRouter_4329327034381
// MI455X (gfx1250) — compile-verified, matrix operand order checked by reading, not run on hardware
//
#include <hip/hip_runtime.h>
#include <hip/hip_bf16.h>

typedef __attribute__((ext_vector_type(16))) _Float16 v16h;
typedef __attribute__((ext_vector_type(8)))  _Float16 v8h;
typedef __attribute__((ext_vector_type(8)))  float    v8f;

#define T_    4
#define B_    16
#define C_    384
#define HW_   784
#define E_    8
#define G_    64          // T_*B_
#define TOK_  50176       // G_*HW_
#define CAP_  196
#define D_    78675968LL  // TOK_*E_*CAP_

// ---------------- zero-fill the two giant one-hot tensors ----------------
__global__ void zero_kernel(float4* __restrict__ p, long n4) {
    long i = (long)blockIdx.x * blockDim.x + threadIdx.x;
    long stride = (long)gridDim.x * blockDim.x;
    float4 z; z.x = 0.f; z.y = 0.f; z.z = 0.f; z.w = 0.f;
    for (; i < n4; i += stride) p[i] = z;
}

// loss slot = 0, capacity slot = 196, zero the per-(g,e) raw-gate sums
__global__ void init_kernel(float* __restrict__ lossp, float* __restrict__ sum_raw) {
    int tid = blockIdx.x * blockDim.x + threadIdx.x;
    if (tid < G_ * E_) sum_raw[tid] = 0.f;
    if (tid == 0) { lossp[0] = 0.f; lossp[1] = (float)CAP_; }
}

// ---------------- LIF scan over T, emit f16 spikes [token, C] ----------------
__global__ void lif_kernel(const float* __restrict__ x, _Float16* __restrict__ spikes) {
    long i = (long)blockIdx.x * blockDim.x + threadIdx.x;
    const long total = (long)B_ * C_ * HW_;
    if (i >= total) return;
    int c  = (int)(i % C_);
    long r = i / C_;
    int hw = (int)(r % HW_);
    int b  = (int)(r / HW_);
    const long stride_t = (long)B_ * C_ * HW_;
    const long xbase = ((long)b * C_ + c) * HW_ + hw;
    float v = 0.f;
    #pragma unroll
    for (int t = 0; t < T_; ++t) {
        float xt = x[(long)t * stride_t + xbase];
        v = v + (xt - v) * 0.5f;               // charge, TAU = 2
        float s = (v >= 1.0f) ? 1.0f : 0.0f;   // fire (V_TH = 1)
        v = v * (1.0f - s);                    // hard reset
        long token = ((long)t * B_ + b) * HW_ + hw;
        spikes[token * C_ + c] = (_Float16)s;
    }
}

// ---------------- pack conv_w [8,384] -> f16 padded to [16,384] ----------------
__global__ void pack_w_kernel(const float* __restrict__ w, _Float16* __restrict__ w16) {
    int i = blockIdx.x * blockDim.x + threadIdx.x;
    if (i >= 16 * C_) return;
    int e = i / C_, c = i % C_;
    w16[i] = (e < E_) ? (_Float16)w[e * C_ + c] : (_Float16)0.0f;
}

// ---------------- WMMA GEMM + BN + softmax + top-2 routing ----------------
__global__ __launch_bounds__(256) void router_gemm(
    const _Float16* __restrict__ sp, const _Float16* __restrict__ w16,
    const float* __restrict__ conv_b, const float* __restrict__ gamma,
    const float* __restrict__ beta,  const float* __restrict__ mean,
    const float* __restrict__ var,
    int* __restrict__ idx1, int* __restrict__ idx2,
    float* __restrict__ g1o, float* __restrict__ g2o,
    float* __restrict__ sum_raw)
{
    __shared__ float tile[8][16][17];           // [wave][token][expert], padded
    const int lane = threadIdx.x & 31;
    const int wv   = threadIdx.x >> 5;
    const int tileId = blockIdx.x * 8 + wv;     // 3136 tiles of 16 tokens
    const int token0 = tileId * 16;
    const int rowm = lane & 15;
    const int hi   = lane >> 4;

    // 16-bit A 16x32 layout: lanes 0-15 hold K{0..7,16..23}, lanes 16-31 K{8..15,24..31}
    const _Float16* Ap = sp  + (size_t)(token0 + rowm) * C_ + hi * 8;
    const _Float16* Bp = w16 + (size_t)rowm * C_ + hi * 8;   // column e = rowm

    v8f acc = {};
    #pragma unroll
    for (int ks = 0; ks < 12; ++ks) {
        const int kb = ks * 32;
        __builtin_prefetch((const void*)(Ap + kb + 64), 0, 0);
        v8h alo = *(const v8h*)(Ap + kb);
        v8h ahi = *(const v8h*)(Ap + kb + 16);
        v8h blo = *(const v8h*)(Bp + kb);
        v8h bhi = *(const v8h*)(Bp + kb + 16);
        v16h a = __builtin_shufflevector(alo, ahi, 0,1,2,3,4,5,6,7,8,9,10,11,12,13,14,15);
        v16h b = __builtin_shufflevector(blo, bhi, 0,1,2,3,4,5,6,7,8,9,10,11,12,13,14,15);
        acc = __builtin_amdgcn_wmma_f32_16x16x32_f16(
                  false, a, false, b, (short)0, acc, false, false);
    }

    // C/D layout: lane l, vgpr r -> (token M = r + 8*(l>>4), expert N = l&15)
    #pragma unroll
    for (int r = 0; r < 8; ++r) tile[wv][r + 8 * hi][rowm] = acc[r];
    __syncthreads();

    if (lane < 16) {
        const int token = token0 + lane;
        float p[E_];
        float m = -1e30f;
        #pragma unroll
        for (int e = 0; e < E_; ++e) {
            float sc = gamma[e] * rsqrtf(var[e] + 1e-5f);
            float y = (tile[wv][lane][e] + conv_b[e] - mean[e]) * sc + beta[e];
            p[e] = y;
            m = fmaxf(m, y);
        }
        float s = 0.f;
        #pragma unroll
        for (int e = 0; e < E_; ++e) { p[e] = __expf(p[e] - m); s += p[e]; }
        float inv = 1.0f / s;
        #pragma unroll
        for (int e = 0; e < E_; ++e) p[e] *= inv;

        int i1 = 0; float v1 = p[0];
        #pragma unroll
        for (int e = 1; e < E_; ++e) if (p[e] > v1) { v1 = p[e]; i1 = e; }
        int i2 = -1; float v2 = -1.f;
        #pragma unroll
        for (int e = 0; e < E_; ++e) if (e != i1 && p[e] > v2) { v2 = p[e]; i2 = e; }

        float denom = v1 + v2 + 1e-9f;
        idx1[token] = i1;
        idx2[token] = i2;
        g1o[token]  = v1 / denom;
        g2o[token]  = v2 / denom;

        const int g = token / HW_;
        #pragma unroll
        for (int e = 0; e < E_; ++e) atomicAdd(&sum_raw[g * E_ + e], p[e]);
    }
}

// ---------------- per-group prefix scan + capacity + scatter ----------------
__global__ __launch_bounds__(32) void scan_scatter(
    const int* __restrict__ idx1, const int* __restrict__ idx2,
    const float* __restrict__ g1, const float* __restrict__ g2,
    const float* __restrict__ sum_raw,
    float* __restrict__ dispatch, float* __restrict__ combine,
    float* __restrict__ lossp)
{
    const int g = blockIdx.x;
    const int lane = threadIdx.x;
    const long gtok = (long)g * HW_;
    const unsigned lt = (1u << lane) - 1u;

    int base1[E_];
    #pragma unroll
    for (int e = 0; e < E_; ++e) base1[e] = 0;

    // pass 1: top-1 positions + scatter
    for (int ch = 0; ch < 25; ++ch) {
        int n = ch * 32 + lane;
        bool valid = n < HW_;
        int e1 = valid ? idx1[gtok + n] : -1;
        int pos1 = CAP_;
        #pragma unroll
        for (int e = 0; e < E_; ++e) {
            unsigned bm = (unsigned)__ballot(e1 == e);
            if (e1 == e) pos1 = base1[e] + __popc(bm & lt);
            base1[e] += __popc(bm);
        }
        if (valid && pos1 < CAP_) {
            long off = ((gtok + n) * E_ + e1) * CAP_ + pos1;
            combine[off]  = g1[gtok + n];
            dispatch[off] = 1.0f;
        }
    }

    // auxiliary loss: mean(density_proxy * density) * E*E
    if (lane == 0) {
        float part = 0.f;
        #pragma unroll
        for (int e = 0; e < E_; ++e) part += sum_raw[g * E_ + e] * (float)base1[e];
        atomicAdd(lossp, part * (1.0f / (784.0f * 784.0f * 8.0f)));
    }

    int base2[E_];
    #pragma unroll
    for (int e = 0; e < E_; ++e) base2[e] = (base1[e] < CAP_) ? base1[e] : CAP_;

    // pass 2: top-2 positions offset by kept top-1 counts
    for (int ch = 0; ch < 25; ++ch) {
        int n = ch * 32 + lane;
        bool valid = n < HW_;
        int e2 = valid ? idx2[gtok + n] : -1;
        int pos2 = CAP_;
        #pragma unroll
        for (int e = 0; e < E_; ++e) {
            unsigned bm = (unsigned)__ballot(e2 == e);
            if (e2 == e) pos2 = base2[e] + __popc(bm & lt);
            base2[e] += __popc(bm);
        }
        if (valid && pos2 < CAP_) {
            long off = ((gtok + n) * E_ + e2) * CAP_ + pos2;
            combine[off]  = g2[gtok + n];
            dispatch[off] = 1.0f;
        }
    }
}

extern "C" void kernel_launch(void* const* d_in, const int* in_sizes, int n_in,
                              void* d_out, int out_size, void* d_ws, size_t ws_size,
                              hipStream_t stream) {
    const float* x      = (const float*)d_in[0];
    const float* conv_w = (const float*)d_in[1];
    const float* conv_b = (const float*)d_in[2];
    const float* gamma  = (const float*)d_in[3];
    const float* beta   = (const float*)d_in[4];
    const float* mean   = (const float*)d_in[5];
    const float* var    = (const float*)d_in[6];

    float* out      = (float*)d_out;
    float* dispatch = out;
    float* combine  = out + D_;
    float* lossp    = out + 2 * D_;   // [loss, capacity]

    char* ws = (char*)d_ws;
    _Float16* spikes = (_Float16*)ws;                       // 50176*384 f16
    _Float16* w16    = spikes + (size_t)TOK_ * C_;          // 16*384 f16
    int*   idx1 = (int*)(w16 + 16 * C_);
    int*   idx2 = idx1 + TOK_;
    float* g1   = (float*)(idx2 + TOK_);
    float* g2   = g1 + TOK_;
    float* sum_raw = g2 + TOK_;                             // 512 floats

    zero_kernel<<<16384, 256, 0, stream>>>((float4*)out, (2 * D_) / 4);
    init_kernel<<<2, 256, 0, stream>>>(lossp, sum_raw);
    lif_kernel<<<((long)B_ * C_ * HW_ + 255) / 256, 256, 0, stream>>>(x, spikes);
    pack_w_kernel<<<(16 * C_ + 255) / 256, 256, 0, stream>>>(conv_w, w16);
    router_gemm<<<TOK_ / 16 / 8, 256, 0, stream>>>(
        spikes, w16, conv_b, gamma, beta, mean, var, idx1, idx2, g1, g2, sum_raw);
    scan_scatter<<<G_, 32, 0, stream>>>(
        idx1, idx2, g1, g2, sum_raw, dispatch, combine, lossp);
}
